// MultiHeadAttention_54400055771452
// MI455X (gfx1250) — matrix operand order repaired and checked by reading, not compiled, not run on hardware
//
#include <hip/hip_runtime.h>
#include <hip/hip_bf16.h>
#include <cstdint>
#include <cstddef>

// ---- problem constants (match reference) ----
#define BB   2
#define SEQ  1024
#define DMODEL 1024
#define NH   16
#define HD   64          // DQK == DV == 64
#define NREL 65          // 2L+1, L=32
#define BSEQ (BB*SEQ)    // 2048

#define TILEA (128 * 32) // halfs per A buffer
#define TILEB (64 * 32)  // halfs per B buffer

typedef _Float16 h16;
typedef __attribute__((ext_vector_type(8)))  _Float16 v8h;
typedef __attribute__((ext_vector_type(16))) _Float16 v16h;
typedef __attribute__((ext_vector_type(8)))  float    v8f;

// ---------------------------------------------------------------------------
// CDNA5 async global->LDS DMA (ASYNCcnt-tracked), CDNA5 ISA
// vdst VGPR carries the LDS byte address (low 32 bits of generic shared ptr,
// per the LDS aperture mapping); 64-bit global address pair, SADDR=off.
// ---------------------------------------------------------------------------
__device__ inline void async_ld_b128(const h16* lds, const h16* g) {
  uint32_t loff = (uint32_t)(uintptr_t)lds;
  uint64_t gptr = (uint64_t)(uintptr_t)g;
  asm volatile("global_load_async_to_lds_b128 %0, %1, off"
               :: "v"(loff), "v"(gptr) : "memory");
}
__device__ inline void wait_async0() {
  asm volatile("s_wait_asynccnt 0" ::: "memory");
}

// ---- tile staging helpers ----
__device__ inline void issue_tileA_async(const h16* A, int lda, int row0, int k0,
                                         h16* dst, int tid) {
  #pragma unroll
  for (int c = 0; c < 2; ++c) {
    const int chunk = tid + c * 256;       // 512 x 16B chunks (128x32 halfs)
    const int row = chunk >> 2, seg = chunk & 3;
    async_ld_b128(dst + row * 32 + seg * 8,
                  A + (size_t)(row0 + row) * lda + k0 + seg * 8);
  }
}
__device__ inline void issue_tileB_async(const h16* Bm, int ldb, int col0, int k0,
                                         h16* dst, int tid) {
  const int row = tid >> 2, seg = tid & 3;  // 256 x 16B chunks (64x32 halfs)
  async_ld_b128(dst + row * 32 + seg * 8,
                Bm + (size_t)(col0 + row) * ldb + k0 + seg * 8);
}
__device__ inline void load_tileA_f32(const float* A, int lda, int row0, int k0,
                                      int K, float (&r)[16], int tid) {
  const int row = tid >> 1;
  const int kk  = (tid & 1) * 16;
  const float* src = A + (size_t)(row0 + row) * lda + k0 + kk;
  if (k0 + 32 < K) __builtin_prefetch(src + 32, 0, 1);  // global_prefetch_b8
  #pragma unroll
  for (int i = 0; i < 16; ++i) r[i] = src[i];
}
__device__ inline void store_tileA_f16(const float (&r)[16], h16* dst, int tid) {
  const int row = tid >> 1;
  const int kk  = (tid & 1) * 16;
  #pragma unroll
  for (int i = 0; i < 16; ++i) dst[row * 32 + kk + i] = (h16)r[i];
}

// ---------------------------------------------------------------------------
// Shared tiled WMMA GEMM core:  C[M,N] += A[M,K] * Bt[N,K]^T
//   block = 256 threads = 8 waves in 4x2;  block tile 128x64;  wave tile 32x32
//   K-step 32, DOUBLE-BUFFERED LDS: tile t+1 is DMA'd with
//   global_load_async_to_lds_b128 while the WMMAs for tile t run; the wave
//   only waits its own ASYNCcnt after compute, then the workgroup barrier
//   publishes the tiles.  f32 A (attn probs) uses the same software pipeline
//   with VGPR staging + f16 convert.  Fragment layouts per CDNA5 ISA 7.12.2.
// ---------------------------------------------------------------------------
template <bool AF32>
__device__ inline void gemm_core(const void* Aptr, const void* Bptr,
                                 int lda, int ldb, int K,
                                 int row0, int col0,
                                 h16* lA, h16* lB, v8f (&acc)[2][2]) {
  const int tid  = threadIdx.x;
  const int wid  = tid >> 5;
  const int lane = tid & 31;
  const int wr   = wid >> 1;      // 0..3  (wave row in block)
  const int wc   = wid & 1;       // 0..1  (wave col in block)
  const int lr   = lane & 15;
  const int hi   = lane >> 4;     // 0 or 1

  float aR[16];

  // ---- prologue: fill buffer 0 ----
  if (AF32) {
    load_tileA_f32((const float*)Aptr, lda, row0, 0, K, aR, tid);
    store_tileA_f16(aR, lA, tid);
  } else {
    issue_tileA_async((const h16*)Aptr, lda, row0, 0, lA, tid);
  }
  issue_tileB_async((const h16*)Bptr, ldb, col0, 0, lB, tid);
  wait_async0();
  __syncthreads();

  for (int k0 = 0; k0 < K; k0 += 32) {
    const int bufi = (k0 >> 5) & 1;
    h16* curA = lA + bufi * TILEA;
    h16* curB = lB + bufi * TILEB;
    h16* nxtA = lA + (bufi ^ 1) * TILEA;
    h16* nxtB = lB + (bufi ^ 1) * TILEB;
    const bool more = (k0 + 32 < K);

    // ---- kick off tile t+1 (overlaps with WMMA below) ----
    if (more) {
      if (AF32) load_tileA_f32((const float*)Aptr, lda, row0, k0 + 32, K, aR, tid);
      else      issue_tileA_async((const h16*)Aptr, lda, row0, k0 + 32, nxtA, tid);
      issue_tileB_async((const h16*)Bptr, ldb, col0, k0 + 32, nxtB, tid);
    }

    // ---- fragments + WMMA on tile t ----
    v16h afr[2], bfr[2];
    #pragma unroll
    for (int f = 0; f < 2; ++f) {
      const int arow = wr * 32 + f * 16 + lr;
      const h16* ap = curA + arow * 32 + hi * 8;            // 16B aligned
      v8h lo  = *(const v8h*)(ap);                          // K = base .. base+7
      v8h hi8 = *(const v8h*)(ap + 16);                     // K = base+16 .. base+23
      afr[f] = __builtin_shufflevector(lo, hi8, 0,1,2,3,4,5,6,7,8,9,10,11,12,13,14,15);
      const int brow = wc * 32 + f * 16 + lr;
      const h16* bp = curB + brow * 32 + hi * 8;            // 16B aligned
      v8h blo = *(const v8h*)(bp);
      v8h bhi = *(const v8h*)(bp + 16);
      bfr[f] = __builtin_shufflevector(blo, bhi, 0,1,2,3,4,5,6,7,8,9,10,11,12,13,14,15);
    }
    #pragma unroll
    for (int fr = 0; fr < 2; ++fr)
      #pragma unroll
      for (int fc = 0; fc < 2; ++fc)
        acc[fr][fc] = __builtin_amdgcn_wmma_f32_16x16x32_f16(
            false, afr[fr], false, bfr[fc], (short)0, acc[fr][fc], false, false);

    // ---- publish tile t+1 ----
    if (more) {
      if (AF32) store_tileA_f16(aR, nxtA, tid);
      wait_async0();            // own wave's async LDS writes complete
    }
    __syncthreads();            // reads of tile t done; tile t+1 visible
  }
}

// ---------------------------------------------------------------------------
// K0: bulk f32 -> f16 conversion (activations and weights)
// ---------------------------------------------------------------------------
__global__ __launch_bounds__(256)
void k_f32tof16(const float* __restrict__ src, h16* __restrict__ dst, int n) {
  const int i = blockIdx.x * 256 + threadIdx.x;
  if (i < n) dst[i] = (h16)src[i];
}

// ---------------------------------------------------------------------------
// K1: projections  X[2048,1024] @ W^T + b   (mode 0: row-major f16, mode 1: V^T)
// ---------------------------------------------------------------------------
__global__ __launch_bounds__(256)
void k_proj(const h16* __restrict__ X, const h16* __restrict__ W,
            const float* __restrict__ bias, h16* __restrict__ outRow,
            h16* __restrict__ outVt, int mode) {
  __shared__ alignas(32) h16 lA[2 * TILEA];
  __shared__ alignas(32) h16 lB[2 * TILEB];
  const int row0 = blockIdx.y * 128;
  const int col0 = blockIdx.x * 64;
  v8f acc[2][2] = {};
  gemm_core<false>(X, W, DMODEL, DMODEL, DMODEL, row0, col0, lA, lB, acc);

  const int lane = threadIdx.x & 31, wid = threadIdx.x >> 5;
  const int wr = wid >> 1, wc = wid & 1, lr = lane & 15, hi = lane >> 4;
  #pragma unroll
  for (int fr = 0; fr < 2; ++fr)
    #pragma unroll
    for (int fc = 0; fc < 2; ++fc)
      #pragma unroll
      for (int r = 0; r < 8; ++r) {
        const int grow = row0 + wr * 32 + fr * 16 + hi * 8 + r;
        const int gcol = col0 + wc * 32 + fc * 16 + lr;
        const float val = acc[fr][fc][r] + bias[gcol];
        if (mode == 0) {
          outRow[(size_t)grow * DMODEL + gcol] = (h16)val;
        } else {
          const int b = grow >> 10, k = grow & 1023;
          const int h = gcol >> 6,  d = gcol & 63;
          outVt[(((size_t)b * NH + h) * HD + d) * SEQ + k] = (h16)val;
        }
      }
}

// ---------------------------------------------------------------------------
// K2: Qmix[b,m,q,nd] = Q[b,q,nd] * T[n][m]   (n = nd>>6)
// ---------------------------------------------------------------------------
__global__ __launch_bounds__(256)
void k_qmix(const h16* __restrict__ Qh, const float* __restrict__ talking,
            h16* __restrict__ Qmix) {
  const size_t t = (size_t)blockIdx.x * 256 + threadIdx.x;  // < 2^25
  const int nd = (int)(t & 1023);
  const int q  = (int)((t >> 10) & 1023);
  const int m  = (int)((t >> 20) & 15);
  const int b  = (int)(t >> 24);
  const float v = (float)Qh[((size_t)(b * SEQ + q)) * DMODEL + nd] *
                  talking[(nd >> 6) * NH + m];
  Qmix[t] = (h16)v;
}

// ---------------------------------------------------------------------------
// K3: qrelmix[bq, m, j] = sum_n (Q[bq,n,:] . W_rel[j,:]) * T[n][m]
// ---------------------------------------------------------------------------
__global__ __launch_bounds__(256)
void k_qrelmix(const h16* __restrict__ Qh, const float* __restrict__ Wrel,
               const float* __restrict__ talking, float* __restrict__ qrelmix) {
  __shared__ float qrow[DMODEL];
  __shared__ float qrel[NH * NREL];
  const int bq = blockIdx.x;  // b*SEQ + q
  const int tid = threadIdx.x;
  for (int i = tid; i < DMODEL; i += 256)
    qrow[i] = (float)Qh[(size_t)bq * DMODEL + i];
  __syncthreads();
  for (int t = tid; t < NH * NREL; t += 256) {
    const int n = t / NREL, j = t % NREL;
    float s = 0.f;
    #pragma unroll 8
    for (int d = 0; d < HD; ++d) s += qrow[n * HD + d] * Wrel[j * HD + d];
    qrel[t] = s;
  }
  __syncthreads();
  for (int t = tid; t < NH * NREL; t += 256) {
    const int m = t / NREL, j = t % NREL;
    float s = 0.f;
    #pragma unroll
    for (int n = 0; n < NH; ++n) s += qrel[n * NREL + j] * talking[n * NH + m];
    qrelmix[(size_t)bq * (NH * NREL) + t] = s;
  }
}

// ---------------------------------------------------------------------------
// K4: scores[b,m] = ( Qmix[b,m] @ Kall[b]^T + qrelmix + |rel|*mixslope[m] )/8, mask
// ---------------------------------------------------------------------------
__global__ __launch_bounds__(256)
void k_scores(const h16* __restrict__ Qmix, const h16* __restrict__ Kall,
              const float* __restrict__ qrelmix, const float* __restrict__ talking,
              const int* __restrict__ qpos, const int* __restrict__ kvpos,
              const unsigned char* __restrict__ mask, float* __restrict__ scoresOut) {
  __shared__ alignas(32) h16 lA[2 * TILEA];
  __shared__ alignas(32) h16 lB[2 * TILEB];
  const int z = blockIdx.z;            // b*NH + m
  const int b = z >> 4, m = z & 15;
  const int row0 = blockIdx.y * 128;   // q
  const int col0 = blockIdx.x * 64;    // k
  const h16* A = Qmix + (size_t)z * SEQ * DMODEL;
  const h16* Bt = Kall + (size_t)b * SEQ * DMODEL;

  v8f acc[2][2] = {};
  gemm_core<false>(A, Bt, DMODEL, DMODEL, DMODEL, row0, col0, lA, lB, acc);

  // mixslope[m] = sum_n 2^(-(n+1)/2) * T[n][m]
  float ms = 0.f;
  #pragma unroll
  for (int n = 0; n < NH; ++n) ms += exp2f(-0.5f * (float)(n + 1)) * talking[n * NH + m];

  const int lane = threadIdx.x & 31, wid = threadIdx.x >> 5;
  const int wr = wid >> 1, wc = wid & 1, lr = lane & 15, hi = lane >> 4;
  #pragma unroll
  for (int fr = 0; fr < 2; ++fr)
    #pragma unroll
    for (int fc = 0; fc < 2; ++fc)
      #pragma unroll
      for (int r = 0; r < 8; ++r) {
        const int q = row0 + wr * 32 + fr * 16 + hi * 8 + r;
        const int k = col0 + wc * 32 + fc * 16 + lr;
        const int rel = qpos[b * SEQ + q] - kvpos[b * SEQ + k];
        int idx = rel; if (idx < -32) idx = -32; if (idx > 32) idx = 32; idx += 32;
        float val = acc[fr][fc][r]
                  + fabsf((float)rel) * ms
                  + qrelmix[(size_t)(b * SEQ + q) * (NH * NREL) + m * NREL + idx];
        val *= 0.125f;  // 1/sqrt(64)
        if (mask[(size_t)b * SEQ * SEQ + (size_t)q * SEQ + k]) val = -10000.0f;
        scoresOut[((size_t)z * SEQ + q) * SEQ + k] = val;
      }
}

// ---------------------------------------------------------------------------
// K5: row softmax (length 1024) + relative-bucket pooling pa[row][65]
// ---------------------------------------------------------------------------
__global__ __launch_bounds__(256)
void k_softmax(const float* __restrict__ scores, const int* __restrict__ qpos,
               const int* __restrict__ kvpos, float* __restrict__ attn,
               float* __restrict__ pa) {
  __shared__ float red[256];
  __shared__ float lpa[NREL];
  const int tid = threadIdx.x;
  const size_t rowi = blockIdx.x;          // z*SEQ + q ; z = b*NH + n
  const int q = (int)(rowi & 1023);
  const int b = (int)(rowi >> 14);
  const float* srow = scores + rowi * SEQ;

  float v[4]; float mx = -3.0e38f;
  #pragma unroll
  for (int e = 0; e < 4; ++e) { v[e] = srow[e * 256 + tid]; mx = fmaxf(mx, v[e]); }
  red[tid] = mx; __syncthreads();
  for (int s = 128; s > 0; s >>= 1) {
    if (tid < s) red[tid] = fmaxf(red[tid], red[tid + s]);
    __syncthreads();
  }
  mx = red[0]; __syncthreads();

  float sum = 0.f;
  #pragma unroll
  for (int e = 0; e < 4; ++e) { v[e] = __expf(v[e] - mx); sum += v[e]; }
  red[tid] = sum; __syncthreads();
  for (int s = 128; s > 0; s >>= 1) {
    if (tid < s) red[tid] += red[tid + s];
    __syncthreads();
  }
  const float inv = 1.0f / red[0];
  if (tid < NREL) lpa[tid] = 0.f;
  __syncthreads();

  const int qp = qpos[b * SEQ + q];
  #pragma unroll
  for (int e = 0; e < 4; ++e) {
    const int k = e * 256 + tid;
    const float a = v[e] * inv;
    attn[rowi * SEQ + k] = a;
    int idx = qp - kvpos[b * SEQ + k];
    if (idx < -32) idx = -32; if (idx > 32) idx = 32; idx += 32;
    atomicAdd(&lpa[idx], a);              // ds_add_f32
  }
  __syncthreads();
  if (tid < NREL) pa[rowi * NREL + tid] = lpa[tid];
}

// ---------------------------------------------------------------------------
// K6: OH[b,q, n*64+d] = attn[b,n] @ V[b,n]  +  pa[b,n,q] @ W_rel
// ---------------------------------------------------------------------------
__global__ __launch_bounds__(256)
void k_attnv(const float* __restrict__ attn, const h16* __restrict__ Vt,
             const float* __restrict__ pa, const float* __restrict__ Wrel,
             h16* __restrict__ OH) {
  __shared__ alignas(32) h16 lA[2 * TILEA];
  __shared__ alignas(32) h16 lB[2 * TILEB];
  const int z = blockIdx.z;            // b*NH + n
  const int b = z >> 4, n = z & 15;
  const int row0 = blockIdx.y * 128;   // q
  const int col0 = 0;                  // d (N = 64)
  const float* A = attn + (size_t)z * SEQ * SEQ;
  const h16* Bt = Vt + (size_t)z * HD * SEQ;

  v8f acc[2][2] = {};
  gemm_core<true>(A, Bt, SEQ, SEQ, SEQ, row0, col0, lA, lB, acc);

  const int lane = threadIdx.x & 31, wid = threadIdx.x >> 5;
  const int wr = wid >> 1, wc = wid & 1, lr = lane & 15, hi = lane >> 4;
  #pragma unroll
  for (int fr = 0; fr < 2; ++fr)
    #pragma unroll
    for (int fc = 0; fc < 2; ++fc)
      #pragma unroll
      for (int r = 0; r < 8; ++r) {
        const int q = row0 + wr * 32 + fr * 16 + hi * 8 + r;
        const int d = col0 + wc * 32 + fc * 16 + lr;
        const float* par = pa + ((size_t)z * SEQ + q) * NREL;
        float pos = 0.f;
        for (int j = 0; j < NREL; ++j) pos += par[j] * Wrel[j * HD + d];
        OH[(size_t)(b * SEQ + q) * DMODEL + n * HD + d] = (h16)(acc[fr][fc][r] + pos);
      }
}

// ---------------------------------------------------------------------------
// K7: out = OH @ W_o^T + b_o   (f32 out)
// ---------------------------------------------------------------------------
__global__ __launch_bounds__(256)
void k_outproj(const h16* __restrict__ OH, const h16* __restrict__ Wo,
               const float* __restrict__ bo, float* __restrict__ out) {
  __shared__ alignas(32) h16 lA[2 * TILEA];
  __shared__ alignas(32) h16 lB[2 * TILEB];
  const int row0 = blockIdx.y * 128;
  const int col0 = blockIdx.x * 64;
  v8f acc[2][2] = {};
  gemm_core<false>(OH, Wo, DMODEL, DMODEL, DMODEL, row0, col0, lA, lB, acc);

  const int lane = threadIdx.x & 31, wid = threadIdx.x >> 5;
  const int wr = wid >> 1, wc = wid & 1, lr = lane & 15, hi = lane >> 4;
  #pragma unroll
  for (int fr = 0; fr < 2; ++fr)
    #pragma unroll
    for (int fc = 0; fc < 2; ++fc)
      #pragma unroll
      for (int r = 0; r < 8; ++r) {
        const int grow = row0 + wr * 32 + fr * 16 + hi * 8 + r;
        const int gcol = col0 + wc * 32 + fc * 16 + lr;
        out[(size_t)grow * DMODEL + gcol] = acc[fr][fc][r] + bo[gcol];
      }
}

// ---------------------------------------------------------------------------
extern "C" void kernel_launch(void* const* d_in, const int* in_sizes, int n_in,
                              void* d_out, int out_size, void* d_ws, size_t ws_size,
                              hipStream_t stream) {
  (void)in_sizes; (void)n_in; (void)out_size; (void)ws_size;
  const float* q_in  = (const float*)d_in[0];
  const float* k_in  = (const float*)d_in[1];
  const float* v_in  = (const float*)d_in[2];
  const unsigned char* mask = (const unsigned char*)d_in[3];
  const int*   qpos  = (const int*)d_in[4];
  const int*   kvpos = (const int*)d_in[5];
  const float* W_q = (const float*)d_in[6];
  const float* b_q = (const float*)d_in[7];
  const float* W_k = (const float*)d_in[8];
  const float* b_k = (const float*)d_in[9];
  const float* W_v = (const float*)d_in[10];
  const float* b_v = (const float*)d_in[11];
  const float* W_o = (const float*)d_in[12];
  const float* b_o = (const float*)d_in[13];
  const float* Wrel = (const float*)d_in[14];
  const float* talking = (const float*)d_in[15];

  // output regions: (out, attn, scores) concatenated, all f32
  float* outP    = (float*)d_out;
  float* attnP   = outP  + (size_t)BB * SEQ * DMODEL;       // 2M
  float* scoresP = attnP + (size_t)BB * NH * SEQ * SEQ;     // +32M

  // workspace carve-up (MiB offsets)
  char* ws = (char*)d_ws;
  h16*   Qh      = (h16*)(ws);                               //  4 MiB
  h16*   Kall    = (h16*)(ws + (size_t)4   * 1048576);       //  4 MiB
  h16*   Vt      = (h16*)(ws + (size_t)8   * 1048576);       //  4 MiB
  h16*   Qmix    = (h16*)(ws + (size_t)12  * 1048576);       // 64 MiB
  float* qrelmix = (float*)(ws + (size_t)76  * 1048576);     // ~8.2 MiB
  float* pa      = (float*)(ws + (size_t)85  * 1048576);     // ~8.2 MiB
  h16*   OH      = (h16*)(ws + (size_t)94  * 1048576);       //  4 MiB
  h16*   Xq16    = (h16*)(ws + (size_t)98  * 1048576);       //  4 MiB
  h16*   Xk16    = (h16*)(ws + (size_t)102 * 1048576);       //  4 MiB
  h16*   Xv16    = (h16*)(ws + (size_t)106 * 1048576);       //  4 MiB
  h16*   Wq16    = (h16*)(ws + (size_t)110 * 1048576);       //  2 MiB
  h16*   Wk16    = (h16*)(ws + (size_t)112 * 1048576);       //  2 MiB
  h16*   Wv16    = (h16*)(ws + (size_t)114 * 1048576);       //  2 MiB
  h16*   Wo16    = (h16*)(ws + (size_t)116 * 1048576);       //  2 MiB

  const dim3 blk(256, 1, 1);

  // f32 -> f16 staging copies
  const int nAct = BSEQ * DMODEL;    // 2,097,152
  const int nW   = DMODEL * DMODEL;  // 1,048,576
  k_f32tof16<<<dim3(nAct / 256), blk, 0, stream>>>(q_in, Xq16, nAct);
  k_f32tof16<<<dim3(nAct / 256), blk, 0, stream>>>(k_in, Xk16, nAct);
  k_f32tof16<<<dim3(nAct / 256), blk, 0, stream>>>(v_in, Xv16, nAct);
  k_f32tof16<<<dim3(nW / 256),   blk, 0, stream>>>(W_q, Wq16, nW);
  k_f32tof16<<<dim3(nW / 256),   blk, 0, stream>>>(W_k, Wk16, nW);
  k_f32tof16<<<dim3(nW / 256),   blk, 0, stream>>>(W_v, Wv16, nW);
  k_f32tof16<<<dim3(nW / 256),   blk, 0, stream>>>(W_o, Wo16, nW);

  const dim3 gFull(16, 16, 1);                               // 1024/64 x 2048/128
  k_proj<<<gFull, blk, 0, stream>>>(Xq16, Wq16, b_q, Qh,   nullptr, 0);
  k_proj<<<gFull, blk, 0, stream>>>(Xk16, Wk16, b_k, Kall, nullptr, 0);
  k_proj<<<gFull, blk, 0, stream>>>(Xv16, Wv16, b_v, nullptr, Vt,   1);

  k_qmix<<<dim3(131072, 1, 1), blk, 0, stream>>>(Qh, talking, Qmix);
  k_qrelmix<<<dim3(BSEQ, 1, 1), blk, 0, stream>>>(Qh, Wrel, talking, qrelmix);

  k_scores<<<dim3(16, 8, BB * NH), blk, 0, stream>>>(Qmix, Kall, qrelmix, talking,
                                                     qpos, kvpos, mask, scoresP);
  k_softmax<<<dim3(BB * NH * SEQ, 1, 1), blk, 0, stream>>>(scoresP, qpos, kvpos,
                                                           attnP, pa);
  k_attnv<<<dim3(1, 8, BB * NH), blk, 0, stream>>>(attnP, Vt, pa, Wrel, OH);
  k_outproj<<<gFull, blk, 0, stream>>>(OH, Wo16, b_o, outP);
}
